// ECInteract_45140106281377
// MI455X (gfx1250) — matrix operand order repaired and checked by reading, not compiled, not run on hardware
//
#include <hip/hip_runtime.h>
#include <hip/hip_bf16.h>
#include <math.h>

// ---------------- types ----------------
typedef __bf16 bf16_t;
typedef __attribute__((ext_vector_type(8)))  __bf16 v8bf;
typedef __attribute__((ext_vector_type(16))) __bf16 v16bf;
typedef __attribute__((ext_vector_type(8)))  float  v8f;

#define CNUM 256
#define FH   256
#define FW   512
#define HW   (FH * FW)          // 131072

// =====================================================================
// 1) NCHW f32 -> NHWC bf16 transpose (LDS tiled, 32x32 tiles)
// =====================================================================
__global__ __launch_bounds__(256) void k_nchw_to_nhwc_bf16(
    const float* __restrict__ src, bf16_t* __restrict__ dst)
{
    __shared__ bf16_t tile[32][33];
    int pblk = blockIdx.x * 32;   // pixel tile base  (4096 blocks)
    int cblk = blockIdx.y * 32;   // channel tile base (8 blocks)
    int tx = threadIdx.x & 31;
    int ty = threadIdx.x >> 5;    // 0..7
#pragma unroll
    for (int j = 0; j < 32; j += 8) {
        int c = cblk + ty + j;
        tile[ty + j][tx] = (bf16_t)src[(size_t)c * HW + pblk + tx];
    }
    __syncthreads();
#pragma unroll
    for (int j = 0; j < 32; j += 8) {
        int p = pblk + ty + j;
        dst[(size_t)p * CNUM + cblk + tx] = tile[tx][ty + j];
    }
}

// =====================================================================
// 2) Weight repack: Wf f32 [cout][cin][3][3] -> Wt bf16 [cout][tap][cin]
// =====================================================================
__global__ __launch_bounds__(256) void k_weights(
    const float* __restrict__ Wf, bf16_t* __restrict__ Wt)
{
    int i = blockIdx.x * 256 + threadIdx.x;
    if (i >= CNUM * CNUM * 9) return;
    int tap  = i % 9;
    int cin  = (i / 9) % CNUM;
    int cout = i / (9 * CNUM);
    Wt[((size_t)cout * 9 + tap) * CNUM + cin] = (bf16_t)Wf[i];
}

// =====================================================================
// 3) cube2equi gather -> aux, stored NHWC bf16
//    faces: 0=back(b) 1=down(d) 5=up(u); 2,3,4 are zero
// =====================================================================
__global__ __launch_bounds__(256) void k_cube2equi(
    const float* __restrict__ fb, const float* __restrict__ fd,
    const float* __restrict__ fu, bf16_t* __restrict__ aux)
{
    const float PI = 3.14159265358979323846f;
    const int h = FH;
    int p = blockIdx.x * 256 + threadIdx.x;
    if (p >= HW) return;
    int y = p >> 9;          // /512
    int x = p & 511;

    float theta = (x + 0.5f) / (float)FW * (2.0f * PI) - PI;
    float phi   = 0.5f * PI - (y + 0.5f) / (float)FH * PI;
    float st = sinf(theta), ct = cosf(theta);
    float sp = sinf(phi),   cp = cosf(phi);
    float dx = cp * st, dy = sp, dz = cp * ct;
    float ax = fabsf(dx), ay = fabsf(dy), az = fabsf(dz);
    float mx = fmaxf(fmaxf(ax, ay), az);
    float pxn = dx / mx, pyn = dy / mx, pzn = dz / mx;

    bool cz = (az >= ax) && (az >= ay);
    bool cx = (!cz) && (ax >= ay);
    int face = cz ? (dz > 0.f ? 2 : 0)
                  : (cx ? (dx > 0.f ? 4 : 3) : (dy > 0.f ? 5 : 1));
    float a  = cz ? (dz > 0.f ? pxn : -pxn)
                  : (cx ? (dx > 0.f ? -pzn : pzn) : pxn);
    float bc = (cz || cx) ? -pyn : (dy > 0.f ? pzn : -pzn);

    float uu = fminf(fmaxf((a  + 1.0f) * 0.5f * (h - 1), 0.0f), (float)(h - 1));
    float vv = fminf(fmaxf((bc + 1.0f) * 0.5f * (h - 1), 0.0f), (float)(h - 1));
    int u0 = (int)floorf(uu), v0 = (int)floorf(vv);
    int u1 = min(u0 + 1, h - 1), v1 = min(v0 + 1, h - 1);
    float wu = uu - (float)u0, wv = vv - (float)v0;
    float w00 = (1.f - wv) * (1.f - wu), w01 = (1.f - wv) * wu;
    float w10 = wv * (1.f - wu),         w11 = wv * wu;

    const float* fp = (face == 0) ? fb : (face == 1) ? fd : (face == 5) ? fu : nullptr;
    bf16_t* op = aux + (size_t)p * CNUM;
    if (fp) {
        size_t o00 = (size_t)v0 * h + u0, o01 = (size_t)v0 * h + u1;
        size_t o10 = (size_t)v1 * h + u0, o11 = (size_t)v1 * h + u1;
        for (int c = 0; c < CNUM; ++c) {
            size_t cb = (size_t)c * h * h;
            float v = w00 * fp[cb + o00] + w01 * fp[cb + o01]
                    + w10 * fp[cb + o10] + w11 * fp[cb + o11];
            op[c] = (bf16_t)v;
        }
    } else {
        for (int c = 0; c < CNUM; ++c) op[c] = (bf16_t)0.0f;
    }
}

// =====================================================================
// 4) 3x3 conv C=256->256, bias+ReLU, implicit GEMM on WMMA bf16.
//    Block = 8 waves -> 128 cout x 64 px. Wave w owns cout tile w and
//    all 4 pixel tiles (A fragment reused 4x, 4 independent acc chains).
//    Double-buffered LDS staging of the halo'd activation tile
//    (3 rows x 66 px x 32 ch, zero-padded, 80B slot stride = conflict-
//    free ds_load_b128). Next chunk's global loads are issued before
//    the current chunk's 36 WMMAs so VMEM overlaps matrix compute;
//    one barrier per chunk. Tap loop is force-unrolled so all DS
//    offsets are immediates and ky/kx are compile-time constants.
// =====================================================================
#define SLOT_STRIDE 40     // bf16 elements per pixel slot (32 data + 8 pad)
#define NSLOT       (3 * 66)

__global__ __launch_bounds__(256) void k_conv3x3_wmma(
    const bf16_t* __restrict__ x0, const bf16_t* __restrict__ x1,
    const bf16_t* __restrict__ Wt, const float* __restrict__ bias,
    float* __restrict__ M0, float* __restrict__ M1)
{
    __shared__ __align__(16) bf16_t btile[2][NSLOT * SLOT_STRIDE]; // 2 x 15840 B

    const bf16_t* __restrict__ xin  = (blockIdx.z == 0) ? x0 : x1;
    float* __restrict__        Mout = (blockIdx.z == 0) ? M0 : M1;

    int wave = threadIdx.x >> 5;                   // 0..7
    int lane = threadIdx.x & 31;
    int p_base    = blockIdx.x * 64;               // 64-px slab (never crosses rows)
    int py        = p_base >> 9;
    int x0pix     = p_base & 511;
    int cout_base = blockIdx.y * 128 + wave * 16;

    int lm = lane & 15;
    int kA = (lane >= 16) ? 16 : 0;                // A-fragment K base (contiguous K)
    int kB = (lane >= 16) ? 16 : 0;                // B-fragment K base

    // ---- staging role: one halo slot per thread (slots 0..197)
    int slot = threadIdx.x;
    bool sactive = slot < NSLOT;
    int hr = slot / 66;                            // halo row 0..2 (sy = py+hr-1)
    int hj = slot % 66;                            // halo col 0..65 (sx = x0-1+hj)
    int sy = py + hr - 1;
    int sx = x0pix + hj - 1;
    bool svalid = sactive && (sy >= 0) && (sy < FH) && (sx >= 0) && (sx < FW);
    size_t goff = svalid ? ((size_t)(sy * FW + sx) * CNUM) : 0;
    const bf16_t* gslot = xin + goff;

    v8f acc[4] = {{}, {}, {}, {}};
    const bf16_t* Abase = Wt + (size_t)(cout_base + lm) * 9 * CNUM + kA;
    const float4 fz = {0.f, 0.f, 0.f, 0.f};

    // per-lane LDS read base (element units): col = lm, +kB within slot
    const bf16_t* ldsrd = &btile[0][(size_t)lm * SLOT_STRIDE + kB];

    // ---- prologue: stage chunk 0 into buffer 0
    {
        float4 p0 = fz, p1 = fz, p2 = fz, p3 = fz;
        if (svalid) {
            const float4* g = (const float4*)gslot;
            p0 = g[0]; p1 = g[1]; p2 = g[2]; p3 = g[3];
        }
        if (sactive) {
            float4* l = (float4*)(&btile[0][(size_t)slot * SLOT_STRIDE]);
            l[0] = p0; l[1] = p1; l[2] = p2; l[3] = p3;
        }
    }
    __syncthreads();

    for (int cc = 0; cc < 8; ++cc) {
        int cin0 = cc * 32;

        // ---- issue next chunk's global loads now (overlap with WMMAs)
        float4 n0 = fz, n1 = fz, n2 = fz, n3 = fz;
        if (cc < 7 && svalid) {
            const float4* g = (const float4*)(gslot + (cc + 1) * 32);
            n0 = g[0]; n1 = g[1]; n2 = g[2]; n3 = g[3];
        }

        const bf16_t* buf = ldsrd + (cc & 1) * (NSLOT * SLOT_STRIDE);

        // ---- 9 taps x 4 pixel tiles on current buffer (fully unrolled:
        //      immediate ds offsets, constant ky/kx)
#pragma unroll
        for (int tap = 0; tap < 9; ++tap) {
            const int ky = tap / 3;                // halo row (constant)
            const int kx = tap % 3;                // halo col shift (constant)
            const bf16_t* Atap = Abase + tap * CNUM + cin0;
            v8bf a0 = *(const v8bf*)(Atap);
            v8bf a1 = *(const v8bf*)(Atap + 8);
            v16bf a = __builtin_shufflevector(a0, a1,
                0, 1, 2, 3, 4, 5, 6, 7, 8, 9, 10, 11, 12, 13, 14, 15);
            v16bf bf4[4];
#pragma unroll
            for (int t = 0; t < 4; ++t) {
                // halo col of this lane's pixel = t*16 + lm + kx (lm folded in base)
                bf4[t] = *(const v16bf*)(buf +
                    (size_t)(ky * 66 + t * 16 + kx) * SLOT_STRIDE);
            }
#pragma unroll
            for (int t = 0; t < 4; ++t) {
                acc[t] = __builtin_amdgcn_wmma_f32_16x16x32_bf16(
                    false, a, false, bf4[t], (short)0, acc[t], false, false);
            }
        }

        // ---- store next chunk into the other buffer, one barrier per chunk
        if (cc < 7) {
            if (sactive) {
                float4* l = (float4*)(&btile[(cc + 1) & 1][(size_t)slot * SLOT_STRIDE]);
                l[0] = n0; l[1] = n1; l[2] = n2; l[3] = n3;
            }
            __syncthreads();
        }
    }

    // ---- epilogue: bias + ReLU, store NCHW f32
    // D layout: VGPR r -> lanes 0-15: (M=r, N=lane), lanes 16-31: (M=8+r, N=lane-16)
    int cbase = cout_base + ((lane >= 16) ? 8 : 0);
#pragma unroll
    for (int t = 0; t < 4; ++t) {
        int pdst = p_base + t * 16 + lm;
#pragma unroll
        for (int r = 0; r < 8; ++r) {
            int cr = cbase + r;
            float v = acc[t][r] + bias[cr];
            v = v > 0.0f ? v : 0.0f;
            Mout[(size_t)cr * HW + pdst] = v;
        }
    }
}

// =====================================================================
// 5) mask = sigmoid( Wm . concat(M_main, M_aux) + bm ) per pixel
// =====================================================================
__global__ __launch_bounds__(256) void k_mask(
    const float* __restrict__ Mm, const float* __restrict__ Ma,
    const float* __restrict__ Wm, const float* __restrict__ bm,
    float* __restrict__ mask)
{
    int p = blockIdx.x * 256 + threadIdx.x;
    if (p >= HW) return;
    float s = bm[0];
    for (int c = 0; c < CNUM; ++c)
        s += Wm[c] * Mm[(size_t)c * HW + p];
    for (int c = 0; c < CNUM; ++c)
        s += Wm[CNUM + c] * Ma[(size_t)c * HW + p];
    mask[p] = 1.0f / (1.0f + expf(-s));
}

// =====================================================================
// 6) out = m + mask * M_aux   (all NCHW f32, float4 vectorized)
// =====================================================================
__global__ __launch_bounds__(256) void k_combine(
    const float* __restrict__ m, const float* __restrict__ Ma,
    const float* __restrict__ mask, float* __restrict__ out)
{
    int i = blockIdx.x * 256 + threadIdx.x;           // over N/4
    const int N4 = (CNUM * HW) / 4;
    if (i >= N4) return;
    float4 mv = ((const float4*)m)[i];
    float4 av = ((const float4*)Ma)[i];
    float4 kv = ((const float4*)mask)[i % (HW / 4)];
    float4 o;
    o.x = mv.x + kv.x * av.x;
    o.y = mv.y + kv.y * av.y;
    o.z = mv.z + kv.z * av.z;
    o.w = mv.w + kv.w * av.w;
    ((float4*)out)[i] = o;
}

// =====================================================================
// launch
// =====================================================================
extern "C" void kernel_launch(void* const* d_in, const int* in_sizes, int n_in,
                              void* d_out, int out_size, void* d_ws, size_t ws_size,
                              hipStream_t stream) {
    const float* m  = (const float*)d_in[0];   // [1,256,256,512]
    const float* fb = (const float*)d_in[1];   // b (back,  face 0)
    const float* fu = (const float*)d_in[2];   // u (up,    face 5)
    const float* fd = (const float*)d_in[3];   // d (down,  face 1)
    const float* Wf = (const float*)d_in[4];   // [256,256,3,3]
    const float* bf = (const float*)d_in[5];   // [256]
    const float* Wm = (const float*)d_in[6];   // [512]
    const float* bm = (const float*)d_in[7];   // [1]
    float* out = (float*)d_out;

    // workspace layout (bytes)
    char* ws = (char*)d_ws;
    bf16_t* x_main = (bf16_t*)(ws + 0);                          //  64 MB
    bf16_t* aux    = (bf16_t*)(ws + 67108864);                   //  64 MB
    bf16_t* Wt     = (bf16_t*)(ws + 2ull * 67108864);            // 1.2 MB
    float*  M_main = (float*) (ws + 2ull * 67108864 + 1179648);  // 128 MB
    float*  M_aux  = (float*) (ws + 2ull * 67108864 + 1179648 + 134217728ull);
    float*  mask   = (float*) (ws + 2ull * 67108864 + 1179648 + 2ull * 134217728ull);

    // 1) m: NCHW f32 -> NHWC bf16
    k_nchw_to_nhwc_bf16<<<dim3(HW / 32, CNUM / 32), 256, 0, stream>>>(m, x_main);
    // 2) weight repack -> bf16 [cout][tap][cin]
    k_weights<<<(CNUM * CNUM * 9 + 255) / 256, 256, 0, stream>>>(Wf, Wt);
    // 3) cube2equi gather -> aux NHWC bf16
    k_cube2equi<<<HW / 256, 256, 0, stream>>>(fb, fd, fu, aux);
    // 4) both 3x3 convs (z=0: main, z=1: aux) via WMMA bf16 + LDS staging
    k_conv3x3_wmma<<<dim3(HW / 64, CNUM / 128, 2), 256, 0, stream>>>(
        x_main, aux, Wt, bf, M_main, M_aux);
    // 5) mask
    k_mask<<<HW / 256, 256, 0, stream>>>(M_main, M_aux, Wm, bm, mask);
    // 6) out = m + mask * M_aux
    k_combine<<<(CNUM * HW / 4) / 256, 256, 0, stream>>>(m, M_aux, mask, out);
}
